// CAD_13211319403325
// MI455X (gfx1250) — matrix operand order repaired and checked by reading, not compiled, not run on hardware
//
#include <hip/hip_runtime.h>
#include <math.h>

typedef __attribute__((ext_vector_type(8)))  float  v8f;
typedef __attribute__((ext_vector_type(8)))  __bf16 v8bf;
typedef __attribute__((ext_vector_type(16))) __bf16 v16bf;

#define ROWS   25088      // 8 * 3136
#define CDIM   1792       // K
#define NCENT  2048       // N
#define KTOP   3
#define NU_    1e-3f
#define ALPHA_ 0.1f
#define MTILE  32         // rows per block
#define PADB   1800       // padded LDS row stride in bf16 elems (3600B, 16B aligned)
#define BLOCKT 256        // 8 waves of 32
#define NPW    256        // centroid columns per wave
#define CSTR   774        // candidate-row stride in floats (conflict-free final scan)

#define WS_HDR_BYTES 16384u
#define WS_NEED ((size_t)WS_HDR_BYTES + 2u * (size_t)CDIM * NCENT * sizeof(unsigned short))

// insert val into ascending 6-array t (keeps the 6 smallest seen)
#define INS6(t, val) do { float _v = (val);                                   \
  if (_v < (t)[5]) (t)[5] = _v;                                               \
  { float _a=(t)[4], _b=(t)[5]; if (_b<_a){(t)[4]=_b;(t)[5]=_a;} }            \
  { float _a=(t)[3], _b=(t)[4]; if (_b<_a){(t)[3]=_b;(t)[4]=_a;} }            \
  { float _a=(t)[2], _b=(t)[3]; if (_b<_a){(t)[2]=_b;(t)[3]=_a;} }            \
  { float _a=(t)[1], _b=(t)[2]; if (_b<_a){(t)[1]=_b;(t)[2]=_a;} }            \
  { float _a=(t)[0], _b=(t)[1]; if (_b<_a){(t)[0]=_b;(t)[1]=_a;} }            \
} while (0)

__device__ __forceinline__ unsigned short rne_bf16(float x) {
  unsigned u = __float_as_uint(x);
  unsigned r = u + 0x7fffu + ((u >> 16) & 1u);
  return (unsigned short)(r >> 16);
}
__device__ __forceinline__ float bf16f(unsigned short h) {
  return __uint_as_float(((unsigned)h) << 16);
}

// ws layout: f32[0]=att_sum, f32[1]=rep_sum, f32[2..2+NCENT)=column norms,
// then (if ws large enough) at byte 16384: bf16 Bhi[n][k], Blo[n][k].
__global__ void cad_colnorm(const float* __restrict__ cent, float* __restrict__ ws) {
  int j = blockIdx.x * blockDim.x + threadIdx.x;
  if (j < NCENT) {
    float s = 0.f;
    for (int k = 0; k < CDIM; ++k) {
      float v = cent[(size_t)k * NCENT + j];   // coalesced across j
      s += v * v;
    }
    ws[2 + j] = s;
  }
  if (blockIdx.x == 0 && threadIdx.x < 2) ws[threadIdx.x] = 0.f; // zero loss accumulators
}

// transpose + hi/lo bf16 split of centroids: cent[k][n] f32 -> Bhi/Blo[n][k] bf16
__global__ void cad_prepB(const float* __restrict__ cent,
                          unsigned short* __restrict__ bhi,
                          unsigned short* __restrict__ blo) {
  __shared__ float t[32][33];
  const int nt = blockIdx.x * 32;
  const int kt = blockIdx.y * 32;
  const int lx = threadIdx.x & 31;
  const int ly = threadIdx.x >> 5;           // 32x8 thread tile
  for (int i = ly; i < 32; i += 8)
    t[i][lx] = cent[(size_t)(kt + i) * NCENT + (nt + lx)];   // coalesced in n
  __syncthreads();
  for (int i = ly; i < 32; i += 8) {
    float x = t[lx][i];                       // = cent[kt+lx][nt+i]
    unsigned short h = rne_bf16(x);
    unsigned short l = rne_bf16(x - bf16f(h));
    bhi[(size_t)(nt + i) * CDIM + kt + lx] = h;  // coalesced in k
    blo[(size_t)(nt + i) * CDIM + kt + lx] = l;
  }
}

template <bool PRECB>
__global__ __launch_bounds__(BLOCKT)
void cad_main(const float* __restrict__ embeds, const float* __restrict__ cent,
              const float* __restrict__ rin, float* __restrict__ out,
              float* __restrict__ ws,
              const unsigned short* __restrict__ bhi,
              const unsigned short* __restrict__ blo) {
  extern __shared__ char smemRaw[];
  unsigned short* AhiU = (unsigned short*)smemRaw;        // MTILE * PADB bf16
  unsigned short* AloU = AhiU + MTILE * PADB;             // MTILE * PADB bf16
  float* rn   = (float*)(AloU + MTILE * PADB);            // 32 row norms
  float* red  = rn + 32;                                  // att[32], rep[32]
  float* cnL  = red + 64;                                 // NCENT column norms
  float* cand = (float*)smemRaw;                          // reused after barrier

  const int tid   = threadIdx.x;
  const int w     = tid >> 5;
  const int l     = tid & 31;
  const int laneM = l & 15;
  const int hi    = l >> 4;
  const int m0    = blockIdx.x * MTILE;

  // --- stage A panel with on-the-fly f32 -> bf16 hi/lo split ---
  for (int idx = tid * 4; idx < MTILE * CDIM; idx += BLOCKT * 4) {
    int row = idx / CDIM;
    int col = idx % CDIM;                                 // multiple of 4
    const float4 v = *(const float4*)(embeds + (size_t)(m0 + row) * CDIM + col);
    unsigned short h0 = rne_bf16(v.x), h1 = rne_bf16(v.y);
    unsigned short h2 = rne_bf16(v.z), h3 = rne_bf16(v.w);
    unsigned short l0 = rne_bf16(v.x - bf16f(h0)), l1 = rne_bf16(v.y - bf16f(h1));
    unsigned short l2 = rne_bf16(v.z - bf16f(h2)), l3 = rne_bf16(v.w - bf16f(h3));
    uint2 hv; hv.x = (unsigned)h0 | ((unsigned)h1 << 16); hv.y = (unsigned)h2 | ((unsigned)h3 << 16);
    uint2 lv; lv.x = (unsigned)l0 | ((unsigned)l1 << 16); lv.y = (unsigned)l2 | ((unsigned)l3 << 16);
    *(uint2*)(AhiU + row * PADB + col) = hv;
    *(uint2*)(AloU + row * PADB + col) = lv;
  }
  for (int j = tid; j < NCENT; j += BLOCKT) cnL[j] = ws[2 + j];
  __syncthreads();

  // --- row norms |x|^2 from reconstructed hi+lo (consistent with GEMM operand) ---
  if (tid < MTILE) {
    const unsigned short* hp = AhiU + tid * PADB;
    const unsigned short* lp = AloU + tid * PADB;
    float s = 0.f;
    for (int kk = 0; kk < CDIM; ++kk) {
      float a = bf16f(hp[kk]) + bf16f(lp[kk]);
      s += a * a;
    }
    rn[tid] = s;
  }
  __syncthreads();

  // --- per-lane sorted top-6 of d2 for 16 rows (2 M-subtiles x 8 VGPR rows) ---
  float tp[16][6];
  #pragma unroll
  for (int i = 0; i < 16; ++i)
    #pragma unroll
    for (int j = 0; j < 6; ++j) tp[i][j] = 3.0e38f;

  float rns[16];
  #pragma unroll
  for (int s = 0; s < 2; ++s)
    #pragma unroll
    for (int v = 0; v < 8; ++v) rns[s * 8 + v] = rn[s * 16 + hi * 8 + v];

  // A-fragment per-lane K offsets for 16-bit WMMA: {16hi..16hi+7} and {16hi+8..+7}
  const int off0 = 16 * hi;
  const int off1 = 16 * hi + 8;
  const unsigned short* a0h = AhiU + (size_t)laneM * PADB;
  const unsigned short* a0l = AloU + (size_t)laneM * PADB;
  const unsigned short* a1h = AhiU + (size_t)(16 + laneM) * PADB;
  const unsigned short* a1l = AloU + (size_t)(16 + laneM) * PADB;
  const int nwbase = w * NPW;

  for (int c = 0; c < NPW / 16; ++c) {
    const int ncol  = nwbase + c * 16 + laneM;
    const float cnv = cnL[ncol];
    // one accumulator per split-product: no same-accumulator back-to-back WMMAs
    v8f acc0hh = {}, acc0hl = {}, acc0lh = {};
    v8f acc1hh = {}, acc1hl = {}, acc1lh = {};
    // B-fragment per-lane K is contiguous: K = 16*hi + e (same k order as the A fragment)
    const unsigned short* bhp = PRECB ? (bhi + (size_t)ncol * CDIM + 16 * hi) : nullptr;
    const unsigned short* blp = PRECB ? (blo + (size_t)ncol * CDIM + 16 * hi) : nullptr;
    const float* bfp = PRECB ? nullptr : (cent + ncol);

    #pragma unroll 2
    for (int k = 0; k < CDIM; k += 32) {
      // A fragments: two b128 halves shuffled into v16bf
      v8bf h0a = *(const v8bf*)(a0h + k + off0);
      v8bf h0b = *(const v8bf*)(a0h + k + off1);
      v8bf l0a = *(const v8bf*)(a0l + k + off0);
      v8bf l0b = *(const v8bf*)(a0l + k + off1);
      v8bf h1a = *(const v8bf*)(a1h + k + off0);
      v8bf h1b = *(const v8bf*)(a1h + k + off1);
      v8bf l1a = *(const v8bf*)(a1l + k + off0);
      v8bf l1b = *(const v8bf*)(a1l + k + off1);
      v16bf ah0 = __builtin_shufflevector(h0a, h0b, 0,1,2,3,4,5,6,7,8,9,10,11,12,13,14,15);
      v16bf al0 = __builtin_shufflevector(l0a, l0b, 0,1,2,3,4,5,6,7,8,9,10,11,12,13,14,15);
      v16bf ah1 = __builtin_shufflevector(h1a, h1b, 0,1,2,3,4,5,6,7,8,9,10,11,12,13,14,15);
      v16bf al1 = __builtin_shufflevector(l1a, l1b, 0,1,2,3,4,5,6,7,8,9,10,11,12,13,14,15);

      v16bf bh, bl;
      if (PRECB) {
        bh = *(const v16bf*)(bhp + k);     // contiguous 32B
        bl = *(const v16bf*)(blp + k);
      } else {
        #pragma unroll
        for (int e = 0; e < 16; ++e) {
          float x = bfp[(size_t)(k + 16 * hi + e) * NCENT];
          unsigned short h = rne_bf16(x);
          bh[e] = __builtin_bit_cast(__bf16, h);
          bl[e] = __builtin_bit_cast(__bf16, rne_bf16(x - bf16f(h)));
        }
      }

      // bf16x3 split-GEMM; 6 distinct accumulators -> RAW distance of 6 WMMAs
      acc0hh = __builtin_amdgcn_wmma_f32_16x16x32_bf16(false, ah0, false, bh, (short)0, acc0hh, false, false);
      acc1hh = __builtin_amdgcn_wmma_f32_16x16x32_bf16(false, ah1, false, bh, (short)0, acc1hh, false, false);
      acc0hl = __builtin_amdgcn_wmma_f32_16x16x32_bf16(false, ah0, false, bl, (short)0, acc0hl, false, false);
      acc1hl = __builtin_amdgcn_wmma_f32_16x16x32_bf16(false, ah1, false, bl, (short)0, acc1hl, false, false);
      acc0lh = __builtin_amdgcn_wmma_f32_16x16x32_bf16(false, al0, false, bh, (short)0, acc0lh, false, false);
      acc1lh = __builtin_amdgcn_wmma_f32_16x16x32_bf16(false, al1, false, bh, (short)0, acc1lh, false, false);
    }

    // d2 = |x|^2 + |c|^2 - 2 x.c ; insert into per-lane top-6
    #pragma unroll
    for (int v = 0; v < 8; ++v) {
      float dot0 = acc0hh[v] + (acc0hl[v] + acc0lh[v]);
      float dot1 = acc1hh[v] + (acc1hl[v] + acc1lh[v]);
      float d0 = rns[v]     + cnv - 2.f * dot0;
      float d1 = rns[8 + v] + cnv - 2.f * dot1;
      INS6(tp[v],     d0);
      INS6(tp[8 + v], d1);
    }
  }

  __syncthreads();   // all waves done with A planes -> reuse as candidate buffer

  #pragma unroll
  for (int s = 0; s < 2; ++s) {
    #pragma unroll
    for (int v = 0; v < 8; ++v) {
      int localRow = s * 16 + hi * 8 + v;
      float* dst = &cand[(size_t)localRow * CSTR + (w * 16 + laneM) * 6];
      #pragma unroll
      for (int j = 0; j < 6; ++j) dst[j] = tp[s * 8 + v][j];
    }
  }
  __syncthreads();

  // --- final merge + score + loss partials (one thread per row) ---
  const float rv = rin[0];
  const float r2 = rv * rv;
  if (tid < MTILE) {
    float best[6];
    #pragma unroll
    for (int j = 0; j < 6; ++j) best[j] = 3.0e38f;
    const float* src = &cand[(size_t)tid * CSTR];
    for (int i = 0; i < 128; ++i) {       // 8 waves * 16 lanes
      #pragma unroll
      for (int j = 0; j < 6; ++j) INS6(best, src[i * 6 + j]);
    }
    float d0 = sqrtf(fmaxf(best[0], 0.f));
    float d1 = sqrtf(fmaxf(best[1], 0.f));
    float d2 = sqrtf(fmaxf(best[2], 0.f));
    float w0 = 1.f / (1.f + __expf(d0 - d1) + __expf(d0 - d2));
    out[1 + m0 + tid] = w0 * d0;          // reshape/transpose is layout-identity
    float att = fmaxf(best[0] - r2, 0.f) + fmaxf(best[1] - r2, 0.f) + fmaxf(best[2] - r2, 0.f);
    float srep = r2 - (best[3] + best[4] + best[5]) * (1.f / 3.f)
                    + (best[0] + best[1] + best[2]) * (1.f / 3.f);
    red[tid] = att;
    red[32 + tid] = fmaxf(srep - ALPHA_, 0.f);
  }
  __syncthreads();
  if (tid == 0) {
    float aS = 0.f, rS = 0.f;
    for (int i = 0; i < MTILE; ++i) { aS += red[i]; rS += red[32 + i]; }
    atomicAdd(&ws[0], aS);
    atomicAdd(&ws[1], rS);
  }
}

__global__ void cad_fin(const float* __restrict__ ws, float* __restrict__ out) {
  if (blockIdx.x == 0 && threadIdx.x == 0) {
    float l_att = (1.0f / NU_) * (ws[0] / (float)(ROWS * KTOP));
    float l_rep = (1.0f / NU_) * (ws[1] / (float)ROWS);
    out[0] = l_att + l_rep;
  }
}

extern "C" void kernel_launch(void* const* d_in, const int* in_sizes, int n_in,
                              void* d_out, int out_size, void* d_ws, size_t ws_size,
                              hipStream_t stream) {
  (void)in_sizes; (void)n_in; (void)out_size;
  const float* embeds = (const float*)d_in[0];   // (8, 3136, 1792) f32
  const float* cent   = (const float*)d_in[1];   // (1792, 2048) f32
  const float* rin    = (const float*)d_in[2];   // (1,) f32
  float* out = (float*)d_out;                    // [0]=loss, [1..]=score (8*3136)
  float* ws  = (float*)d_ws;

  const size_t smemBytes =
      (size_t)(2 * MTILE * PADB) * sizeof(unsigned short) +   // bf16 hi/lo A planes
      (size_t)(32 + 64 + NCENT) * sizeof(float);              // rn, red, col norms

  cad_colnorm<<<NCENT / 256, 256, 0, stream>>>(cent, ws);

  if (ws_size >= WS_NEED) {
    unsigned short* bhi = (unsigned short*)((char*)d_ws + WS_HDR_BYTES);
    unsigned short* blo = bhi + (size_t)CDIM * NCENT;
    cad_prepB<<<dim3(NCENT / 32, CDIM / 32), 256, 0, stream>>>(cent, bhi, blo);
    hipFuncSetAttribute(reinterpret_cast<const void*>(&cad_main<true>),
                        hipFuncAttributeMaxDynamicSharedMemorySize, (int)smemBytes);
    cad_main<true><<<ROWS / MTILE, BLOCKT, smemBytes, stream>>>(
        embeds, cent, rin, out, ws, bhi, blo);
  } else {
    hipFuncSetAttribute(reinterpret_cast<const void*>(&cad_main<false>),
                        hipFuncAttributeMaxDynamicSharedMemorySize, (int)smemBytes);
    cad_main<false><<<ROWS / MTILE, BLOCKT, smemBytes, stream>>>(
        embeds, cent, rin, out, ws, nullptr, nullptr);
  }
  cad_fin<<<1, 32, 0, stream>>>(ws, out);
}
